// MutiTaskModel_35974646071338
// MI455X (gfx1250) — compile-verified, matrix operand order checked by reading, not run on hardware
//
#include <hip/hip_runtime.h>
#include <hip/hip_bf16.h>
#include <math.h>

// ---------- types ----------
typedef __attribute__((ext_vector_type(16))) __bf16 v16bf;
typedef __attribute__((ext_vector_type(8)))  float  f32x8;
typedef __attribute__((ext_vector_type(4)))  unsigned int u32x4;
typedef __attribute__((ext_vector_type(8)))  unsigned int u32x8;

#define WMMA_BF16(A, Bm, C) \
    __builtin_amdgcn_wmma_f32_16x16x32_bf16(false, (A), false, (Bm), (short)0, (C), false, false)

#define ZERO8 {0.f,0.f,0.f,0.f,0.f,0.f,0.f,0.f}

// Problem constants
#define BB 32
#define SS 128
#define TT 64
#define EE 256
#define HH 512
#define GG 1536   // 3*H
#define VV 32000

__device__ __forceinline__ float sigmf(float x) { return 1.f / (1.f + __expf(-x)); }

// Load one 16x32 bf16 operand tile from a row-major [rows, ld] bf16 matrix.
// Layout per CDNA5 ISA 7.12.2 (16-bit A/B): lane<16 -> row=lane, K elems
// [k0..k0+7] then [k0+16..k0+23]; lane>=16 -> row=lane-16, [k0+8..15],[k0+24..31].
__device__ __forceinline__ v16bf ldtile(const __bf16* p, int ld, int row0, int k0) {
    int lane = threadIdx.x & 31;
    int half = lane >> 4;
    int lr   = lane & 15;
    const __bf16* q = p + (size_t)(row0 + lr) * ld + k0 + half * 8;
    union { v16bf v; uint4 u[2]; } t;
    t.u[0] = *(const uint4*)(q);
    t.u[1] = *(const uint4*)(q + 16);
    return t.v;
}

// Same tile layout, but source is fp32 (LDS-resident h), converted on the fly.
__device__ __forceinline__ v16bf ldtile_f32(const float* p, int ld, int row0, int k0) {
    int lane = threadIdx.x & 31;
    int half = lane >> 4;
    int lr   = lane & 15;
    const float* q = p + (size_t)(row0 + lr) * ld + k0 + half * 8;
    const float4* q4 = (const float4*)q;
    float4 f0 = q4[0], f1 = q4[1], f2 = q4[4], f3 = q4[5];
    float tmp[16] = {f0.x,f0.y,f0.z,f0.w, f1.x,f1.y,f1.z,f1.w,
                     f2.x,f2.y,f2.z,f2.w, f3.x,f3.y,f3.z,f3.w};
    v16bf r;
    #pragma unroll
    for (int i = 0; i < 16; ++i) r[i] = (__bf16)tmp[i];
    return r;
}

// ---------- Tensor Data Mover: DMA a [64 x 512] bf16 tile (64 KB) into LDS ----------
// D# per CDNA5 ISA: 2D tile, data_size=8B (row = 128 x 8B).
// Issued by one wave; tracked by TENSORcnt.
__device__ __forceinline__ void tdm_load_chunk(const __bf16* gaddr, void* laddr) {
    unsigned long long ga = (unsigned long long)(uintptr_t)gaddr;
    unsigned int la = (unsigned int)(uintptr_t)laddr;  // flat-LDS low bits = LDS byte addr
    u32x4 g0;
    g0[0] = 1u;                                        // count=1, user descriptor
    g0[1] = la;                                        // lds_addr
    g0[2] = (unsigned int)ga;                          // global_addr[31:0]
    g0[3] = (unsigned int)((ga >> 32) & 0x01FFFFFFu)   // global_addr[56:32]
            | (2u << 30);                              // type = 2 ("image")
    u32x8 g1;
    g1[0] = 3u << 16;                                  // workgroup_mask=0, data_size=8B
    g1[1] = (128u & 0xFFFFu) << 16;                    // tensor_dim0 lo16 (128 units)
    g1[2] = (128u >> 16) | (64u << 16);                // tensor_dim0 hi | tensor_dim1 lo (64)
    g1[3] = (64u >> 16) | (128u << 16);                // tensor_dim1 hi | tile_dim0 = 128
    g1[4] = 64u;                                       // tile_dim1 = 64, tile_dim2 = 0
    g1[5] = 128u;                                      // tensor_dim0_stride = 128 (8B units)
    g1[6] = 0u;                                        // stride0 hi | stride1 lo
    g1[7] = 0u;                                        // stride1 hi
    u32x4 gz = {0u, 0u, 0u, 0u};                       // groups 2/3: all dims unused
    asm volatile("tensor_load_to_lds %0, %1, %2, %3"
                 :: "s"(g0), "s"(g1), "s"(gz), "s"(gz)
                 : "memory");
}

// ---------- elementwise helpers ----------
__global__ void cvt_bf16(const float* __restrict__ in, __bf16* __restrict__ out, int n) {
    int i = blockIdx.x * blockDim.x + threadIdx.x;
    if (i < n) out[i] = (__bf16)in[i];
}

// one block per token, 256 threads = E
__global__ void embed_bf16(const int* __restrict__ ids, const float* __restrict__ emb,
                           __bf16* __restrict__ out) {
    int tok = blockIdx.x;
    int e   = threadIdx.x;
    out[(size_t)tok * EE + e] = (__bf16)emb[(size_t)ids[tok] * EE + e];
}

// ---------- generic C[M,N] = A[M,K](bf16) * W[N,K](bf16)^T + bias[N], fp32 out ----------
__global__ void __launch_bounds__(256) wmma_gemm_bias(
        const __bf16* __restrict__ A, const __bf16* __restrict__ W,
        const float* __restrict__ bias, float* __restrict__ C,
        int M, int N, int K) {
    int wtile  = blockIdx.x * 8 + (threadIdx.x >> 5);
    int nT     = N >> 4;
    int ntiles = (M >> 4) * nT;
    if (wtile >= ntiles) return;                 // wave-uniform: EXEC stays all-1s
    int mt = wtile / nT, nt = wtile % nT;
    f32x8 acc = ZERO8;
    for (int k0 = 0; k0 < K; k0 += 32)
        acc = WMMA_BF16(ldtile(A, K, mt * 16, k0), ldtile(W, K, nt * 16, k0), acc);
    int lane = threadIdx.x & 31, half = lane >> 4, lr = lane & 15;
    float bv = bias[nt * 16 + lr];
    #pragma unroll
    for (int v = 0; v < 8; ++v)
        C[(size_t)(mt * 16 + v + 8 * half) * N + nt * 16 + lr] = acc[v] + bv;
}

// ---------- encoder GRU: one persistent workgroup, 16 waves, h in LDS ----------
// gx: [B,S,3H] fp32 (includes bih). Whh: [3H,H] bf16. Gate order r,z,n.
__global__ void __launch_bounds__(512) encoder_gru(
        const float* __restrict__ gx, const __bf16* __restrict__ Whh,
        const float* __restrict__ bhh, const int* __restrict__ lens,
        float* __restrict__ src_ht, __bf16* __restrict__ src_ht_b) {
    __shared__ float hf[BB * HH];                 // 64 KB
    int tid = threadIdx.x;
    for (int i = tid; i < BB * HH; i += 512) hf[i] = 0.f;
    __syncthreads();

    int wv = tid >> 5, lane = tid & 31, half = lane >> 4, lr = lane & 15;
    // 64 units = (m in 0..1) x (jt in 0..31); wave owns u = wv + 16*k
    int lenb[4][8];
    #pragma unroll
    for (int k = 0; k < 4; ++k) {
        int u = wv + 16 * k, m = u & 1;
        #pragma unroll
        for (int v = 0; v < 8; ++v) lenb[k][v] = lens[m * 16 + v + 8 * half];
    }

    float hw[4][8];
    for (int s = 0; s < SS; ++s) {
        #pragma unroll
        for (int k = 0; k < 4; ++k) {
            int u = wv + 16 * k;
            int m = u & 1, jt = u >> 1;
            int r0 = m * 16, j0 = jt * 16;
            f32x8 ar = ZERO8, az = ZERO8, an = ZERO8;
            for (int k0 = 0; k0 < HH; k0 += 32) {
                v16bf a = ldtile_f32(hf, HH, r0, k0);
                ar = WMMA_BF16(a, ldtile(Whh, HH, j0,            k0), ar);
                az = WMMA_BF16(a, ldtile(Whh, HH, HH + j0,       k0), az);
                an = WMMA_BF16(a, ldtile(Whh, HH, 2 * HH + j0,   k0), an);
            }
            int j = j0 + lr;
            float br = bhh[j], bz = bhh[HH + j], bn = bhh[2 * HH + j];
            #pragma unroll
            for (int v = 0; v < 8; ++v) {
                int b = r0 + v + 8 * half;
                const float* g = gx + ((size_t)b * SS + s) * GG;
                float hold = hf[b * HH + j];
                float r = sigmf(g[j]          + ar[v] + br);
                float z = sigmf(g[HH + j]     + az[v] + bz);
                float n = tanhf(g[2 * HH + j] + r * (an[v] + bn));
                float hn = (1.f - z) * n + z * hold;
                hw[k][v] = (s < lenb[k][v]) ? hn : hold;
            }
        }
        __syncthreads();   // all reads of hf done
        #pragma unroll
        for (int k = 0; k < 4; ++k) {
            int u = wv + 16 * k, m = u & 1, jt = u >> 1;
            int j = jt * 16 + lr;
            #pragma unroll
            for (int v = 0; v < 8; ++v)
                hf[(m * 16 + v + 8 * half) * HH + j] = hw[k][v];
        }
        __syncthreads();   // all writes visible
    }
    for (int i = tid; i < BB * HH; i += 512) {
        src_ht[i]   = hf[i];
        src_ht_b[i] = (__bf16)hf[i];
    }
}

// ---------- conditional GRU decoder: same persistent pattern ----------
// gi: [B,T,3H] (includes bi). Wh: [3H,H] bf16. pg: [B,3H] (ctx@Wp^T+bp). Gates r,i,n.
__global__ void __launch_bounds__(512) decoder_gru(
        const float* __restrict__ gi, const __bf16* __restrict__ Wh,
        const float* __restrict__ bh, const float* __restrict__ pg,
        const float* __restrict__ h0, __bf16* __restrict__ trg_h) {
    __shared__ float hf[BB * HH];                 // 64 KB
    int tid = threadIdx.x;
    for (int i = tid; i < BB * HH; i += 512) hf[i] = h0[i];
    __syncthreads();

    int wv = tid >> 5, lane = tid & 31, half = lane >> 4, lr = lane & 15;
    float pr[4][8], pi[4][8], pn[4][8];
    #pragma unroll
    for (int k = 0; k < 4; ++k) {
        int u = wv + 16 * k, m = u & 1, jt = u >> 1;
        int j = jt * 16 + lr;
        #pragma unroll
        for (int v = 0; v < 8; ++v) {
            int b = m * 16 + v + 8 * half;
            const float* p = pg + (size_t)b * GG;
            pr[k][v] = p[j]; pi[k][v] = p[HH + j]; pn[k][v] = p[2 * HH + j];
        }
    }

    float hw[4][8];
    for (int t = 0; t < TT; ++t) {
        #pragma unroll
        for (int k = 0; k < 4; ++k) {
            int u = wv + 16 * k, m = u & 1, jt = u >> 1;
            int r0 = m * 16, j0 = jt * 16;
            f32x8 ar = ZERO8, ai = ZERO8, an = ZERO8;
            for (int k0 = 0; k0 < HH; k0 += 32) {
                v16bf a = ldtile_f32(hf, HH, r0, k0);
                ar = WMMA_BF16(a, ldtile(Wh, HH, j0,          k0), ar);
                ai = WMMA_BF16(a, ldtile(Wh, HH, HH + j0,     k0), ai);
                an = WMMA_BF16(a, ldtile(Wh, HH, 2 * HH + j0, k0), an);
            }
            int j = j0 + lr;
            float br = bh[j], bi2 = bh[HH + j], bn = bh[2 * HH + j];
            #pragma unroll
            for (int v = 0; v < 8; ++v) {
                int b = r0 + v + 8 * half;
                const float* g = gi + ((size_t)b * TT + t) * GG;
                float hold  = hf[b * HH + j];
                float reset = sigmf(g[j]          + (ar[v] + br)  + pr[k][v]);
                float ing   = sigmf(g[HH + j]     + (ai[v] + bi2) + pi[k][v]);
                float newg  = tanhf(g[2 * HH + j] + reset * (an[v] + bn) + pn[k][v]);
                hw[k][v] = newg + ing * (hold - newg);
            }
        }
        __syncthreads();
        #pragma unroll
        for (int k = 0; k < 4; ++k) {
            int u = wv + 16 * k, m = u & 1, jt = u >> 1;
            int j = jt * 16 + lr;
            #pragma unroll
            for (int v = 0; v < 8; ++v) {
                int b = m * 16 + v + 8 * half;
                float hy = hw[k][v];
                hf[b * HH + j] = hy;
                trg_h[((size_t)b * TT + t) * HH + j] = (__bf16)hy;  // row = b*T+t
            }
        }
        __syncthreads();
    }
}

// ---------- fused vocab GEMM + online logsumexp + target logit -> per-row NLL ----------
// Hb: [2048,512] bf16 (row = b*T+t). Wb: [32000,512] bf16.
// One block per 32 rows; W streamed through LDS by the Tensor Data Mover in
// 64-row (64 KB) chunks, double-buffered; 8 waves = 2 Mtiles x 4 Ntiles.
__global__ void __launch_bounds__(256) vocab_nll(
        const __bf16* __restrict__ Hb, const __bf16* __restrict__ Wb,
        const float* __restrict__ bias, const int* __restrict__ tgt,
        float* __restrict__ nll) {
    __shared__ __bf16 wbuf[2][64 * HH];            // 2 x 64 KB
    const int mt2  = blockIdx.x;                   // 0..63 -> rows mt2*32 .. +31
    const int wv   = threadIdx.x >> 5;             // 0..7
    const int mhalf = wv >> 2, nq = wv & 3;        // Mtile half, Ntile-in-chunk
    const int lane = threadIdx.x & 31, half = lane >> 4, lr = lane & 15;

    const int row0 = mt2 * 32 + mhalf * 16;
    // A tiles for this wave's Mtile are invariant: keep all 16 in registers.
    v16bf atile[16];
    #pragma unroll
    for (int kk = 0; kk < 16; ++kk) atile[kk] = ldtile(Hb, HH, row0, kk * 32);

    float mx[8], sm[8], tg[8];
    int tid8[8];
    #pragma unroll
    for (int v = 0; v < 8; ++v) {
        mx[v] = -3.0e38f; sm[v] = 0.f; tg[v] = 0.f;
        tid8[v] = tgt[row0 + v + 8 * half];
    }

    const int NCHUNK = VV / 64;                    // 500 chunks of 64 vocab rows
    if (wv == 0) {                                 // wave-uniform branch
        tdm_load_chunk(Wb, &wbuf[0][0]);
        __builtin_amdgcn_s_wait_tensorcnt(0);
    }
    __syncthreads();

    for (int c = 0; c < NCHUNK; ++c) {
        int buf = c & 1;
        if (wv == 0 && c + 1 < NCHUNK)             // prefetch next chunk via TDM
            tdm_load_chunk(Wb + (size_t)(c + 1) * 64 * HH, &wbuf[buf ^ 1][0]);

        f32x8 acc = ZERO8;
        #pragma unroll
        for (int kk = 0; kk < 16; ++kk)
            acc = WMMA_BF16(atile[kk], ldtile(wbuf[buf], HH, nq * 16, kk * 32), acc);

        int col  = c * 64 + nq * 16 + lr;
        float bv = bias[col];
        #pragma unroll
        for (int v = 0; v < 8; ++v) {
            float x = acc[v] + bv;
            if (col == tid8[v]) tg[v] += x;
            if (x > mx[v]) { sm[v] = sm[v] * __expf(mx[v] - x) + 1.f; mx[v] = x; }
            else           { sm[v] += __expf(x - mx[v]); }
        }

        if (wv == 0 && c + 1 < NCHUNK)
            __builtin_amdgcn_s_wait_tensorcnt(0);  // next buffer landed
        __syncthreads();                           // publish LDS to all waves
    }

    // reduce across the 16 lanes holding one row (xor stays within each half)
    float* pf = (float*)&wbuf[0][0];               // reuse staged-W LDS as scratch
    #pragma unroll
    for (int v = 0; v < 8; ++v) {
        float m = mx[v], s = sm[v], t = tg[v];
        for (int off = 1; off < 16; off <<= 1) {
            float om = __shfl_xor(m, off, 32);
            float os = __shfl_xor(s, off, 32);
            float M2 = fmaxf(m, om);
            s = s * __expf(m - M2) + os * __expf(om - M2);
            m = M2;
            t += __shfl_xor(t, off, 32);
        }
        if (lr == 0) {
            int idx = v + 8 * half;                // row within Mtile
            pf[wv * 16 + idx]       = m;
            pf[128 + wv * 16 + idx] = s;
            pf[256 + wv * 16 + idx] = t;
        }
    }
    __syncthreads();
    if (threadIdx.x < 32) {                        // one thread per block row
        int r  = threadIdx.x;                      // row within block
        int mq = r >> 4, ri = r & 15;
        int w0 = mq * 4;                           // 4 waves share this Mtile
        float m = pf[w0 * 16 + ri], s = pf[128 + w0 * 16 + ri], t = pf[256 + w0 * 16 + ri];
        for (int q = 1; q < 4; ++q) {
            int w = w0 + q;
            float om = pf[w * 16 + ri], os = pf[128 + w * 16 + ri];
            float M2 = fmaxf(m, om);
            s = s * __expf(m - M2) + os * __expf(om - M2);
            m = M2;
            t += pf[256 + w * 16 + ri];
        }
        nll[mt2 * 32 + r] = (m + logf(s)) - t;     // logsumexp - target_logit
    }
}

__global__ void __launch_bounds__(256) reduce_mean(const float* __restrict__ nll,
                                                   float* __restrict__ out) {
    __shared__ float red[256];
    int tid = threadIdx.x;
    float s = 0.f;
    for (int i = tid; i < BB * TT; i += 256) s += nll[i];
    red[tid] = s;
    __syncthreads();
    for (int st = 128; st > 0; st >>= 1) {
        if (tid < st) red[tid] += red[tid + st];
        __syncthreads();
    }
    if (tid == 0) out[0] = red[0] / (float)(BB * TT);
}

// ---------- host ----------
extern "C" void kernel_launch(void* const* d_in, const int* in_sizes, int n_in,
                              void* d_out, int out_size, void* d_ws, size_t ws_size,
                              hipStream_t stream) {
    (void)in_sizes; (void)n_in; (void)out_size; (void)ws_size;
    const int*   src     = (const int*)  d_in[0];
    const int*   src_len = (const int*)  d_in[1];
    const int*   trg_in  = (const int*)  d_in[2];
    const int*   trg_out = (const int*)  d_in[3];
    const float* src_emb = (const float*)d_in[4];
    const float* trg_emb = (const float*)d_in[5];
    const float* eWih    = (const float*)d_in[6];
    const float* eWhh    = (const float*)d_in[7];
    const float* ebih    = (const float*)d_in[8];
    const float* ebhh    = (const float*)d_in[9];
    const float* dWi     = (const float*)d_in[10];
    const float* dbi     = (const float*)d_in[11];
    const float* dWh     = (const float*)d_in[12];
    const float* dbh     = (const float*)d_in[13];
    const float* dWp     = (const float*)d_in[14];
    const float* dbp     = (const float*)d_in[15];
    const float* oW      = (const float*)d_in[16];
    const float* ob      = (const float*)d_in[17];

    char* base = (char*)d_ws;
    size_t off = 0;
    auto carve = [&](size_t bytes) -> char* {
        off = (off + 255) & ~(size_t)255;
        char* p = base + off;
        off += bytes;
        return p;
    };

    __bf16* src_e    = (__bf16*)carve((size_t)BB * SS * EE * 2);
    __bf16* trg_e    = (__bf16*)carve((size_t)BB * TT * EE * 2);
    __bf16* eWih_b   = (__bf16*)carve((size_t)GG * EE * 2);
    __bf16* eWhh_b   = (__bf16*)carve((size_t)GG * HH * 2);
    __bf16* dWi_b    = (__bf16*)carve((size_t)GG * EE * 2);
    __bf16* dWh_b    = (__bf16*)carve((size_t)GG * HH * 2);
    __bf16* dWp_b    = (__bf16*)carve((size_t)GG * HH * 2);
    __bf16* oW_b     = (__bf16*)carve((size_t)VV * HH * 2);
    float*  gx       = (float*) carve((size_t)BB * SS * GG * 4);
    float*  gi       = (float*) carve((size_t)BB * TT * GG * 4);
    float*  src_ht   = (float*) carve((size_t)BB * HH * 4);
    __bf16* src_ht_b = (__bf16*)carve((size_t)BB * HH * 2);
    float*  pgate    = (float*) carve((size_t)BB * GG * 4);
    __bf16* trg_hb   = (__bf16*)carve((size_t)BB * TT * HH * 2);
    float*  nll      = (float*) carve((size_t)BB * TT * 4);

    auto grid1d = [](int n) { return (n + 255) / 256; };

    // weight conversions fp32 -> bf16
    cvt_bf16<<<grid1d(GG * EE), 256, 0, stream>>>(eWih, eWih_b, GG * EE);
    cvt_bf16<<<grid1d(GG * HH), 256, 0, stream>>>(eWhh, eWhh_b, GG * HH);
    cvt_bf16<<<grid1d(GG * EE), 256, 0, stream>>>(dWi,  dWi_b,  GG * EE);
    cvt_bf16<<<grid1d(GG * HH), 256, 0, stream>>>(dWh,  dWh_b,  GG * HH);
    cvt_bf16<<<grid1d(GG * HH), 256, 0, stream>>>(dWp,  dWp_b,  GG * HH);
    cvt_bf16<<<grid1d(VV * HH), 256, 0, stream>>>(oW,   oW_b,   VV * HH);

    // embedding gathers (bf16)
    embed_bf16<<<BB * SS, EE, 0, stream>>>(src,    src_emb, src_e);
    embed_bf16<<<BB * TT, EE, 0, stream>>>(trg_in, trg_emb, trg_e);

    // input-gate precompute GEMMs (include input bias)
    {   // gx: M=4096, N=1536, K=256
        int tiles = (BB * SS / 16) * (GG / 16);
        wmma_gemm_bias<<<(tiles + 7) / 8, 256, 0, stream>>>(src_e, eWih_b, ebih, gx,
                                                            BB * SS, GG, EE);
    }
    {   // gi: M=2048, N=1536, K=256
        int tiles = (BB * TT / 16) * (GG / 16);
        wmma_gemm_bias<<<(tiles + 7) / 8, 256, 0, stream>>>(trg_e, dWi_b, dbi, gi,
                                                            BB * TT, GG, EE);
    }

    // encoder recurrence (persistent single workgroup)
    encoder_gru<<<1, 512, 0, stream>>>(gx, eWhh_b, ebhh, src_len, src_ht, src_ht_b);

    // projected-context gates: pgate = src_ht @ Wp^T + bp  (M=32, N=1536, K=512)
    {
        int tiles = (BB / 16) * (GG / 16);
        wmma_gemm_bias<<<(tiles + 7) / 8, 256, 0, stream>>>(src_ht_b, dWp_b, dbp, pgate,
                                                            BB, GG, HH);
    }

    // decoder recurrence
    decoder_gru<<<1, 512, 0, stream>>>(gi, dWh_b, dbh, pgate, src_ht, trg_hb);

    // fused vocab GEMM (TDM-staged W) + log-softmax NLL
    vocab_nll<<<(BB * TT) / 32, 256, 0, stream>>>(trg_hb, oW_b, ob, trg_out, nll);

    // mean over B*T rows
    reduce_mean<<<1, 256, 0, stream>>>(nll, (float*)d_out);
}
